// FusionMamba55_8641474199636
// MI455X (gfx1250) — compile-verified, matrix operand order checked by reading, not run on hardware
//
#include <hip/hip_runtime.h>
#include <hip/hip_bf16.h>
#include <math.h>

// ---------------- problem constants ----------------
#define BB       2
#define DMODEL   256
#define HH       48
#define WW       48
#define HWSZ     (HH*WW)          // 2304
#define LSEQ     (2*HWSZ)         // 4608 per batch
#define NROWS    (BB*LSEQ)        // 9216 total rows
#define DINNER   512
#define DSTATE   16
#define DCONV    4
#define DTRANK   16
#define CHUNK    256
#define NCHUNK   (LSEQ/CHUNK)     // 18

typedef _Float16 h8  __attribute__((ext_vector_type(8)));
typedef _Float16 v16h __attribute__((ext_vector_type(16)));
typedef float    v8f  __attribute__((ext_vector_type(8)));

// ---------------- WMMA fragment load (16-bit A/B layout, CDNA5 §7.12.2) ---
// lane&15 = row (M for A, N for B); per-lane 16 halfs cover K=0..31 as
// [k0+hi*8 .. +7] and [k0+16+hi*8 .. +7]  (hi = lane>>4)
__device__ inline v16h load_frag16(const _Float16* base, int ld, int k0, int lane) {
    int r  = lane & 15;
    int hi = lane >> 4;
    const _Float16* p = base + (size_t)r * ld + k0 + hi * 8;
    h8 lo = *(const h8*)(p);
    h8 up = *(const h8*)(p + 16);
    v16h f;
#pragma unroll
    for (int i = 0; i < 8; ++i) { f[i] = lo[i]; f[i + 8] = up[i]; }
    return f;
}

__device__ inline v16h frag_from_lds(const _Float16* p /*16B aligned*/) {
    h8 lo = *(const h8*)(p);
    h8 up = *(const h8*)(p + 16);
    v16h f;
#pragma unroll
    for (int i = 0; i < 8; ++i) { f[i] = lo[i]; f[i + 8] = up[i]; }
    return f;
}

// ============ 64x64-tile WMMA GEMM with LDS-staged B ======================
//  C[M,N] = A[M,K] * Bw[N,K]^T ; requires M%64==0, N%64==0, K%32==0
//  block(32,4): wave w owns M-subtile w; B tile (64 rows x 32 K) staged in
//  LDS once per K-step and shared by all 4 waves; each wave reuses one A
//  fragment against 4 B fragments -> 4 WMMAs per A load.
//  mode 0: plain   mode 1: + bias[col]   mode 2: + skip[0]*ep[row*N+col]
__global__ void __launch_bounds__(128)
gemm_wmma_64x64(const _Float16* __restrict__ A,
                const _Float16* __restrict__ Bw,
                float* __restrict__ C,
                int M, int N, int K, int mode,
                const float* __restrict__ ep,
                const float* __restrict__ skip) {
    __shared__ __align__(16) _Float16 Bs[64 * 32];   // 4 KB
    int lane = threadIdx.x;                // 0..31 (wave32)
    int wid  = threadIdx.y;                // 0..3
    int t    = wid * 32 + lane;            // 0..127
    int tn = blockIdx.x, tm = blockIdx.y;
    const _Float16* Ab = A  + (size_t)(tm * 64 + wid * 16) * K;
    const _Float16* Bb = Bw + (size_t)tn * 64 * K;

    int brow = t >> 1, bpart = t & 1;      // 64 rows x 2 half-rows
    const _Float16* bsrc = Bb + (size_t)brow * K + bpart * 16;
    _Float16* bdst = &Bs[brow * 32 + bpart * 16];

    int r = lane & 15, hi = lane >> 4;
    v8f acc0 = {}, acc1 = {}, acc2 = {}, acc3 = {};

    for (int k0 = 0; k0 < K; k0 += 32) {
        h8 b0 = *(const h8*)(bsrc + k0);
        h8 b1 = *(const h8*)(bsrc + k0 + 8);
        if (k0 + 32 < K) {
            __builtin_prefetch((const void*)(bsrc + k0 + 32), 0, 0);
            __builtin_prefetch((const void*)(Ab + (size_t)r * K + k0 + 32), 0, 0);
        }
        __syncthreads();                   // previous-iteration readers done
        *(h8*)(bdst)     = b0;
        *(h8*)(bdst + 8) = b1;
        __syncthreads();                   // tile visible to all waves

        v16h a = load_frag16(Ab, K, k0, lane);
        v16h f0 = frag_from_lds(&Bs[(0 * 16 + r) * 32 + hi * 8]);
        v16h f1 = frag_from_lds(&Bs[(1 * 16 + r) * 32 + hi * 8]);
        v16h f2 = frag_from_lds(&Bs[(2 * 16 + r) * 32 + hi * 8]);
        v16h f3 = frag_from_lds(&Bs[(3 * 16 + r) * 32 + hi * 8]);
        acc0 = __builtin_amdgcn_wmma_f32_16x16x32_f16(false, a, false, f0, (short)0, acc0, false, false);
        acc1 = __builtin_amdgcn_wmma_f32_16x16x32_f16(false, a, false, f1, (short)0, acc1, false, false);
        acc2 = __builtin_amdgcn_wmma_f32_16x16x32_f16(false, a, false, f2, (short)0, acc2, false, false);
        acc3 = __builtin_amdgcn_wmma_f32_16x16x32_f16(false, a, false, f3, (short)0, acc3, false, false);
    }

    v8f accs[4] = {acc0, acc1, acc2, acc3};
#pragma unroll
    for (int j = 0; j < 4; ++j) {
        int col = tn * 64 + j * 16 + r;
#pragma unroll
        for (int rr = 0; rr < 8; ++rr) {
            int row = tm * 64 + wid * 16 + rr + hi * 8;
            float v = accs[j][rr];
            if (mode == 1)      v += ep[col];
            else if (mode == 2) v += skip[0] * ep[(size_t)row * N + col];
            C[(size_t)row * N + col] = v;
        }
    }
}

// ============ 16x16-tile WMMA GEMM (for N not multiple of 64) =============
__global__ void gemm_wmma_f16(const _Float16* __restrict__ A,
                              const _Float16* __restrict__ Bw,
                              float* __restrict__ C,
                              int M, int N, int K, int mode,
                              const float* __restrict__ ep,
                              const float* __restrict__ skip) {
    int lane = threadIdx.x;
    int tn = blockIdx.x, tm = blockIdx.y;
    const _Float16* Ab = A  + (size_t)tm * 16 * K;
    const _Float16* Bb = Bw + (size_t)tn * 16 * K;
    v8f acc = {};
    for (int k0 = 0; k0 < K; k0 += 32) {
        if (k0 + 32 < K) {
            __builtin_prefetch((const void*)(Ab + (size_t)(lane & 15) * K + k0 + 32), 0, 0);
            __builtin_prefetch((const void*)(Bb + (size_t)(lane & 15) * K + k0 + 32), 0, 0);
        }
        v16h a = load_frag16(Ab, K, k0, lane);
        v16h b = load_frag16(Bb, K, k0, lane);
        acc = __builtin_amdgcn_wmma_f32_16x16x32_f16(
                  false, a, false, b, (short)0, acc, false, false);
    }
    int hi = lane >> 4, n = lane & 15;
    int col = tn * 16 + n;
#pragma unroll
    for (int r = 0; r < 8; ++r) {
        int row = tm * 16 + r + hi * 8;
        float v = acc[r];
        if (mode == 1)      v += ep[col];
        else if (mode == 2) v += skip[0] * ep[(size_t)row * N + col];
        C[(size_t)row * N + col] = v;
    }
}

// ---------------- f32 -> f16 convert ----------------
__global__ void cvt_f32_f16(const float* __restrict__ src,
                            _Float16* __restrict__ dst, int n) {
    int i = blockIdx.x * blockDim.x + threadIdx.x;
    if (i < n) dst[i] = (_Float16)src[i];
}

// ---------------- LN over channels, building cat from pan/ms -------------
__global__ void ln_concat(const float* __restrict__ pan, const float* __restrict__ ms,
                          const float* __restrict__ nw, const float* __restrict__ nb,
                          float* __restrict__ cat, _Float16* __restrict__ cat16) {
    __shared__ float s[DMODEL];
    int row = blockIdx.x;              // 0..NROWS-1
    int c = threadIdx.x;               // 0..255
    int b = row / LSEQ, l = row % LSEQ;
    float v;
    if (l < HWSZ) v = pan[((size_t)b * DMODEL + c) * HWSZ + l];
    else          v = ms [((size_t)b * DMODEL + c) * HWSZ + (l - HWSZ)];
    cat[(size_t)row * DMODEL + c] = v;
    s[c] = v; __syncthreads();
    for (int off = DMODEL / 2; off > 0; off >>= 1) {
        if (c < off) s[c] += s[c + off];
        __syncthreads();
    }
    float mu = s[0] / DMODEL; __syncthreads();
    float d = v - mu;
    s[c] = d * d; __syncthreads();
    for (int off = DMODEL / 2; off > 0; off >>= 1) {
        if (c < off) s[c] += s[c + off];
        __syncthreads();
    }
    float rs = rsqrtf(s[0] / DMODEL + 1e-5f);
    cat16[(size_t)row * DMODEL + c] = (_Float16)(d * rs * nw[c] + nb[c]);
}

// ---------------- LN of fusion buffer -> f16 -----------------------------
__global__ void ln_rows(const float* __restrict__ src,
                        const float* __restrict__ nw, const float* __restrict__ nb,
                        _Float16* __restrict__ dst16) {
    __shared__ float s[DMODEL];
    int row = blockIdx.x;
    int c = threadIdx.x;
    float v = src[(size_t)row * DMODEL + c];
    s[c] = v; __syncthreads();
    for (int off = DMODEL / 2; off > 0; off >>= 1) {
        if (c < off) s[c] += s[c + off];
        __syncthreads();
    }
    float mu = s[0] / DMODEL; __syncthreads();
    float d = v - mu;
    s[c] = d * d; __syncthreads();
    for (int off = DMODEL / 2; off > 0; off >>= 1) {
        if (c < off) s[c] += s[c + off];
        __syncthreads();
    }
    float rs = rsqrtf(s[0] / DMODEL + 1e-5f);
    dst16[(size_t)row * DMODEL + c] = (_Float16)(d * rs * nw[c] + nb[c]);
}

// ---------------- causal depthwise conv (width 4) + SiLU -----------------
__global__ void conv_silu(const float* __restrict__ xz,
                          const float* __restrict__ cw, const float* __restrict__ cb,
                          float* __restrict__ u, _Float16* __restrict__ u16) {
    int row = blockIdx.x;              // 0..NROWS-1
    int d = threadIdx.x;               // 0..511
    int b = row / LSEQ, l = row % LSEQ;
    float acc = cb[d];
#pragma unroll
    for (int j = 0; j < DCONV; ++j) {
        int lj = l - (DCONV - 1) + j;
        if (lj >= 0)
            acc += xz[((size_t)b * LSEQ + lj) * (2 * DINNER) + d] * cw[d * DCONV + j];
    }
    float sv = acc / (1.0f + __expf(-acc));
    u  [(size_t)row * DINNER + d] = sv;
    u16[(size_t)row * DINNER + d] = (_Float16)sv;
}

// ---------------- delta = softplus(dt @ Wdt^T + bdt) ---------------------
__global__ void delta_softplus(const float* __restrict__ xdbl,
                               const float* __restrict__ wdt, const float* __restrict__ bdt,
                               float* __restrict__ delta) {
    int row = blockIdx.x;
    int d = threadIdx.x;
    float acc = bdt[d];
#pragma unroll
    for (int r = 0; r < DTRANK; ++r)
        acc += xdbl[(size_t)row * 48 + r] * wdt[d * DTRANK + r];
    float sp = (acc > 20.0f) ? acc : log1pf(__expf(acc));
    delta[(size_t)row * DINNER + d] = sp;
}

// ---------------- selective scan: pass 1 (per-chunk local scan) ----------
__global__ void scan_pass1(const float* __restrict__ delta, const float* __restrict__ xdbl,
                           const float* __restrict__ u, const float* __restrict__ Alog,
                           float* __restrict__ Aprod, float* __restrict__ Hfin) {
    int tid = threadIdx.x;             // n + 16*dl
    int n = tid & 15, dl = tid >> 4;
    int d = blockIdx.x * 16 + dl;
    int chunk = blockIdx.y, b = blockIdx.z;
    float Av = -__expf(Alog[d * DSTATE + n]);
    float h = 0.f, ap = 1.f;
    int rowbase = b * LSEQ + chunk * CHUNK;
    for (int t = 0; t < CHUNK; ++t) {
        int row = rowbase + t;
        float dt = delta[(size_t)row * DINNER + d];
        float a = __expf(dt * Av);
        float bt = xdbl[(size_t)row * 48 + 16 + n];
        float ut = u[(size_t)row * DINNER + d];
        h = a * h + dt * bt * ut;
        ap *= a;
    }
    size_t idx = (((size_t)b * NCHUNK + chunk) * DINNER + d) * DSTATE + n;
    Aprod[idx] = ap; Hfin[idx] = h;
}

// ---------------- scan: pass 2 (sequential combine over 18 chunks) -------
__global__ void scan_pass2(const float* __restrict__ Aprod, const float* __restrict__ Hfin,
                           float* __restrict__ Hinit) {
    int g = blockIdx.x * blockDim.x + threadIdx.x;   // (b,d,n)
    if (g >= BB * DINNER * DSTATE) return;
    int n = g % DSTATE, d = (g / DSTATE) % DINNER, b = g / (DSTATE * DINNER);
    float h = 0.f;
    for (int k = 0; k < NCHUNK; ++k) {
        size_t idx = (((size_t)b * NCHUNK + k) * DINNER + d) * DSTATE + n;
        Hinit[idx] = h;
        h = Aprod[idx] * h + Hfin[idx];
    }
}

// ---------------- scan: pass 3 (apply init state, gate, -> f16) ----------
__global__ void scan_pass3(const float* __restrict__ delta, const float* __restrict__ xdbl,
                           const float* __restrict__ u, const float* __restrict__ xz,
                           const float* __restrict__ Alog, const float* __restrict__ Dv,
                           const float* __restrict__ Hinit,
                           _Float16* __restrict__ g16) {
    int tid = threadIdx.x;
    int n = tid & 15, dl = tid >> 4;
    int d = blockIdx.x * 16 + dl;
    int chunk = blockIdx.y, b = blockIdx.z;
    float Av = -__expf(Alog[d * DSTATE + n]);
    size_t sidx = (((size_t)b * NCHUNK + chunk) * DINNER + d) * DSTATE + n;
    float h = Hinit[sidx];
    float Dd = Dv[d];
    int rowbase = b * LSEQ + chunk * CHUNK;
    for (int t = 0; t < CHUNK; ++t) {
        int row = rowbase + t;
        float dt = delta[(size_t)row * DINNER + d];
        float a = __expf(dt * Av);
        float bt = xdbl[(size_t)row * 48 + 16 + n];
        float ut = u[(size_t)row * DINNER + d];
        h = a * h + dt * bt * ut;
        float ct = xdbl[(size_t)row * 48 + 32 + n];
        float v = h * ct;
        // reduce over the 16 state lanes (wave32 -> two 16-lane groups)
        v += __shfl_xor(v, 8, 16);
        v += __shfl_xor(v, 4, 16);
        v += __shfl_xor(v, 2, 16);
        v += __shfl_xor(v, 1, 16);
        if (n == 0) {
            float y = v + ut * Dd;
            float z = xz[(size_t)row * (2 * DINNER) + DINNER + d];
            float g = y * (z / (1.0f + __expf(-z)));
            g16[(size_t)row * DINNER + d] = (_Float16)g;
        }
    }
}

// ---------------- final recombine: pan part + ms part --------------------
__global__ void combine_out(const float* __restrict__ f2, float* __restrict__ out) {
    int i = blockIdx.x * blockDim.x + threadIdx.x;   // (b*C + c)*HW + l
    if (i >= BB * DMODEL * HWSZ) return;
    int l = i % HWSZ;
    int c = (i / HWSZ) % DMODEL;
    int b = i / (HWSZ * DMODEL);
    float a = f2[((size_t)b * LSEQ + l) * DMODEL + c];
    float m = f2[((size_t)b * LSEQ + HWSZ + l) * DMODEL + c];
    out[i] = a + m;
}

// ==========================================================================
extern "C" void kernel_launch(void* const* d_in, const int* in_sizes, int n_in,
                              void* d_out, int out_size, void* d_ws, size_t ws_size,
                              hipStream_t stream) {
    const float* pan       = (const float*)d_in[0];
    const float* ms        = (const float*)d_in[1];
    const float* norm_w    = (const float*)d_in[2];
    const float* norm_b    = (const float*)d_in[3];
    const float* out_w     = (const float*)d_in[4];
    const float* out_b     = (const float*)d_in[5];
    const float* skip      = (const float*)d_in[6];
    const float* in_proj_w = (const float*)d_in[7];
    const float* conv_w    = (const float*)d_in[8];
    const float* conv_b    = (const float*)d_in[9];
    const float* x_proj_w  = (const float*)d_in[10];
    const float* dt_proj_w = (const float*)d_in[11];
    const float* dt_proj_b = (const float*)d_in[12];
    const float* A_log     = (const float*)d_in[13];
    const float* Dvec      = (const float*)d_in[14];
    const float* mo_w      = (const float*)d_in[15];
    float* outp = (float*)d_out;

    // ---- workspace carve-up (256B aligned) ----
    char* base = (char*)d_ws;
    size_t off = 0;
    auto alloc = [&](size_t bytes) {
        char* p = base + off;
        off = (off + bytes + 255) & ~(size_t)255;
        return p;
    };
    float*    cat    = (float*)   alloc((size_t)NROWS * DMODEL * 4);
    _Float16* cat16  = (_Float16*)alloc((size_t)NROWS * DMODEL * 2); // reused for LN2 f16
    _Float16* win16  = (_Float16*)alloc((size_t)2 * DINNER * DMODEL * 2);
    _Float16* wx16   = (_Float16*)alloc((size_t)48 * DINNER * 2);
    _Float16* wmo16  = (_Float16*)alloc((size_t)DMODEL * DINNER * 2);
    _Float16* wout16 = (_Float16*)alloc((size_t)DMODEL * DMODEL * 2);
    float*    xz     = (float*)   alloc((size_t)NROWS * 2 * DINNER * 4);
    float*    u      = (float*)   alloc((size_t)NROWS * DINNER * 4);
    _Float16* u16    = (_Float16*)alloc((size_t)NROWS * DINNER * 2);
    float*    xdbl   = (float*)   alloc((size_t)NROWS * 48 * 4);
    float*    delta  = (float*)   alloc((size_t)NROWS * DINNER * 4);
    float*    Aprod  = (float*)   alloc((size_t)BB * NCHUNK * DINNER * DSTATE * 4);
    float*    Hfin   = (float*)   alloc((size_t)BB * NCHUNK * DINNER * DSTATE * 4);
    float*    Hinit  = (float*)   alloc((size_t)BB * NCHUNK * DINNER * DSTATE * 4);
    _Float16* g16    = (_Float16*)alloc((size_t)NROWS * DINNER * 2);
    float*    fusion = (float*)   alloc((size_t)NROWS * DMODEL * 4);
    float*    f2     = (float*)   alloc((size_t)NROWS * DMODEL * 4);
    (void)ws_size; (void)in_sizes; (void)n_in; (void)out_size;

    // 1) cat + LN -> f16
    ln_concat<<<NROWS, DMODEL, 0, stream>>>(pan, ms, norm_w, norm_b, cat, cat16);

    // 2) weight conversions f32 -> f16
    {
        int n1 = 2 * DINNER * DMODEL;
        cvt_f32_f16<<<(n1 + 255) / 256, 256, 0, stream>>>(in_proj_w, win16, n1);
        int n2 = 48 * DINNER;
        cvt_f32_f16<<<(n2 + 255) / 256, 256, 0, stream>>>(x_proj_w, wx16, n2);
        int n3 = DMODEL * DINNER;
        cvt_f32_f16<<<(n3 + 255) / 256, 256, 0, stream>>>(mo_w, wmo16, n3);
        int n4 = DMODEL * DMODEL;
        cvt_f32_f16<<<(n4 + 255) / 256, 256, 0, stream>>>(out_w, wout16, n4);
    }

    // 3) in_proj GEMM: (9216x256) x (1024x256)^T -> xz    [WMMA 64x64+LDS]
    gemm_wmma_64x64<<<dim3((2 * DINNER) / 64, NROWS / 64), dim3(32, 4), 0, stream>>>(
        cat16, win16, xz, NROWS, 2 * DINNER, DMODEL, 0, nullptr, nullptr);

    // 4) causal depthwise conv + SiLU -> u (f32 + f16)
    conv_silu<<<NROWS, DINNER, 0, stream>>>(xz, conv_w, conv_b, u, u16);

    // 5) x_proj GEMM: (9216x512) x (48x512)^T -> xdbl      [WMMA 16x16]
    gemm_wmma_f16<<<dim3(48 / 16, NROWS / 16), 32, 0, stream>>>(
        u16, wx16, xdbl, NROWS, 48, DINNER, 0, nullptr, nullptr);

    // 6) delta = softplus(dt @ Wdt^T + b)
    delta_softplus<<<NROWS, DINNER, 0, stream>>>(xdbl, dt_proj_w, dt_proj_b, delta);

    // 7) chunked selective scan (3 passes)
    scan_pass1<<<dim3(DINNER / 16, NCHUNK, BB), 256, 0, stream>>>(
        delta, xdbl, u, A_log, Aprod, Hfin);
    {
        int tot = BB * DINNER * DSTATE;
        scan_pass2<<<(tot + 255) / 256, 256, 0, stream>>>(Aprod, Hfin, Hinit);
    }
    scan_pass3<<<dim3(DINNER / 16, NCHUNK, BB), 256, 0, stream>>>(
        delta, xdbl, u, xz, A_log, Dvec, Hinit, g16);

    // 8) mamba_out GEMM + skip*cat epilogue -> fusion      [WMMA 64x64+LDS]
    gemm_wmma_64x64<<<dim3(DMODEL / 64, NROWS / 64), dim3(32, 4), 0, stream>>>(
        g16, wmo16, fusion, NROWS, DMODEL, DINNER, 2, cat, skip);

    // 9) LN(fusion) -> f16 (reuse cat16 buffer)
    ln_rows<<<NROWS, DMODEL, 0, stream>>>(fusion, norm_w, norm_b, cat16);

    // 10) out GEMM + bias -> f2                            [WMMA 64x64+LDS]
    gemm_wmma_64x64<<<dim3(DMODEL / 64, NROWS / 64), dim3(32, 4), 0, stream>>>(
        cat16, wout16, f2, NROWS, DMODEL, DMODEL, 1, out_b, nullptr);

    // 11) pan part + ms part -> output (2,256,48,48)
    {
        int tot = BB * DMODEL * HWSZ;
        combine_out<<<(tot + 255) / 256, 256, 0, stream>>>(f2, outp);
    }
}
